// ResNet_19524921328220
// MI455X (gfx1250) — compile-verified, matrix operand order checked by reading, not run on hardware
//
#include <hip/hip_runtime.h>
#include <math.h>

// ---------------------------------------------------------------------------
// MI455X (gfx1250, wave32) spherical-mesh ResNet.
// Core op: z[N, cout] = gather7(x)[N, 7*cin] @ W[7*cin, cout] + b
// as WMMA bf16 GEMM with f32 accumulation (v_wmma_f32_16x16x32_bf16).
// - weights pre-packed once per layer in B-fragment order (bf16)
// - activations shadowed in bf16 so the 7x-amplified gather moves half bytes
// - neighbor table staged in LDS once per block
// - K loop software-pipelined with ping-pong LDS A buffers
// ---------------------------------------------------------------------------

typedef __attribute__((ext_vector_type(16))) __bf16 v16bf;
typedef __attribute__((ext_vector_type(8)))  float  v8f;

__device__ __forceinline__ __bf16 f2bf(float f) {
    union { float f; unsigned u; } v; v.f = f;
    unsigned r = (v.u + 0x7FFFu + ((v.u >> 16) & 1u)) >> 16;   // RNE
    unsigned short s = (unsigned short)r;
    __bf16 out;
    __builtin_memcpy(&out, &s, 2);
    return out;
}

// Fragment index mapping (16-bit 16x32 operand, ISA 7.12.2):
//   k(0..31) = b*16 + hi*8 + t, with b=k>>4, hi=(k>>3)&1, t=k&7
//   vector elem e = (b*4 + t/2)*2 + (t&1); lane = hi*16 + (row/col & 15)
__device__ __forceinline__ void frag_pos(int kk32, int& hi, int& e) {
    int b = kk32 >> 4, t = kk32 & 7;
    hi = (kk32 >> 3) & 1;
    e  = ((b << 2) + (t >> 1)) * 2 + (t & 1);
}

// ---------------------------------------------------------------------------
// cvt_w: W[K, cout] f32 -> Wf bf16 packed per (col-tile ct, k-chunk kc):
//   Wf[((ct*nkc + kc)*32 + lane)*16 + e]
// ---------------------------------------------------------------------------
__global__ __launch_bounds__(256)
void cvt_w(const float* __restrict__ W, __bf16* __restrict__ Wf,
           int K, int cout)
{
    int idx = blockIdx.x * 256 + threadIdx.x;
    if (idx >= K * cout) return;
    int k = idx / cout, col = idx - k * cout;
    int ct = col >> 4, nl = col & 15;
    int kc = k >> 5,  kk = k & 31;
    int hi, e; frag_pos(kk, hi, e);
    int nkc = K >> 5;
    Wf[((ct * nkc + kc) * 32 + ((hi << 4) + nl)) * 16 + e] = f2bf(W[idx]);
}

// cvt_x: f32 -> bf16 elementwise (for the network input only)
__global__ __launch_bounds__(256)
void cvt_x(const float* __restrict__ in, __bf16* __restrict__ out, int tot)
{
    int idx = blockIdx.x * 256 + threadIdx.x;
    if (idx < tot) out[idx] = f2bf(in[idx]);
}

// ---------------------------------------------------------------------------
// conv_wmma: block = 32-node M tile (2 WMMA M-subtiles), 8 waves each own a
// 16-col c_out tile. K in chunks of 64 (2 WMMA K-subchunks). Ping-pong LDS:
// gather for chunk i+1 is issued before the WMMAs of chunk i.
// ---------------------------------------------------------------------------
__global__ __launch_bounds__(256)
void conv_wmma(const __bf16* __restrict__ xb, const int* __restrict__ neigh,
               const __bf16* __restrict__ Wf, const float* __restrict__ bias,
               float* __restrict__ z, int N, int cinShift, int cout)
{
    __shared__ alignas(32) __bf16 As[2][4 * 32 * 16];   // ping-pong, 8 KB
    __shared__ int nbr_s[32 * 7];

    const int cin   = 1 << cinShift;
    const int K     = 7 * cin;
    const int nkc   = K >> 5;
    const int nch   = K >> 6;
    const int mbase = blockIdx.x * 32;
    const int tid   = threadIdx.x;
    const int wave  = tid >> 5;
    const int lane  = tid & 31;
    const int ct    = blockIdx.y * 8 + wave;
    const bool active = (ct * 16) < cout;   // cout is a multiple of 16
    const int hi = lane >> 4;
    const int nl = lane & 15;

    // Stage neighbor table once (clamped for OOB rows; their values are
    // never stored, so any finite garbage is fine).
    for (int idx = tid; idx < 224; idx += 256) {
        int mm = idx / 7, j = idx - mm * 7;
        int nd = mbase + mm;
        nbr_s[idx] = (nd < N) ? neigh[nd * 7 + j] : 0;
    }

    // Per-slot loop-invariant gather coordinates + LDS fragment destinations.
    int slot_mm[8], slot_kk[8], slot_dst[8];
#pragma unroll
    for (int s = 0; s < 8; ++s) {
        int e = tid + s * 256;
        int mm = e >> 6, kk = e & 63;
        slot_mm[s] = mm; slot_kk[s] = kk;
        int mt = mm >> 4, m = mm & 15;
        int sub = kk >> 5, kk32 = kk & 31;
        int hh, el; frag_pos(kk32, hh, el);
        slot_dst[s] = (((mt << 1) + sub) * 32 + ((hh << 4) + m)) * 16 + el;
    }

    __syncthreads();    // nbr_s ready

    v8f acc0 = {}, acc1 = {};
    __bf16 vals[8];

    auto gather = [&](int k0) {
#pragma unroll
        for (int s = 0; s < 8; ++s) {
            int k = k0 + slot_kk[s];
            int j = k >> cinShift;
            int c = k & (cin - 1);
            vals[s] = xb[(nbr_s[slot_mm[s] * 7 + j] << cinShift) + c];
        }
    };

    gather(0);

    for (int i = 0; i < nch; ++i) {
        int p = i & 1;
        __bf16* Ap = &As[p][0];
#pragma unroll
        for (int s = 0; s < 8; ++s) Ap[slot_dst[s]] = vals[s];
        __syncthreads();
        if (i + 1 < nch) gather((i + 1) * 64);   // loads fly during WMMAs
        if (active) {
#pragma unroll
            for (int sub = 0; sub < 2; ++sub) {
                int kc = i * 2 + sub;
                v16bf b  = *(const v16bf*)&Wf[((ct * nkc + kc) * 32 + lane) * 16];
                v16bf a0 = *(const v16bf*)&Ap[((0 + sub) * 32 + lane) * 16];
                v16bf a1 = *(const v16bf*)&Ap[((2 + sub) * 32 + lane) * 16];
                acc0 = __builtin_amdgcn_wmma_f32_16x16x32_bf16(
                           false, a0, false, b, (short)0, acc0, false, false);
                acc1 = __builtin_amdgcn_wmma_f32_16x16x32_bf16(
                           false, a1, false, b, (short)0, acc1, false, false);
            }
        }
        __syncthreads();    // buf p free for overwrite in iteration i+2
    }

    if (active) {
        int col = ct * 16 + nl;
        float bv = bias[col];
#pragma unroll
        for (int r = 0; r < 8; ++r) {
            int row0 = mbase + r + hi * 8;
            if (row0 < N) z[row0 * cout + col] = acc0[r] + bv;
            int row1 = mbase + 16 + r + hi * 8;
            if (row1 < N) z[row1 * cout + col] = acc1[r] + bv;
        }
    }
}

// ---------------------------------------------------------------------------
// BN statistics: per-channel sum / sumsq. sums[0..1023]=sum, [1024..2047]=sq.
// ---------------------------------------------------------------------------
__global__ __launch_bounds__(256)
void bn_stats(const float* __restrict__ z, float* __restrict__ sums,
              int N, int C)
{
    int r0 = blockIdx.x * 128;
    int rend = r0 + 128; if (rend > N) rend = N;
    for (int c = threadIdx.x; c < C; c += 256) {
        float s = 0.f, q = 0.f;
        for (int r = r0; r < rend; ++r) {
            float v = z[r * C + c];
            s += v; q += v * v;
        }
        atomicAdd(&sums[c], s);
        atomicAdd(&sums[1024 + c], q);
    }
}

// ---------------------------------------------------------------------------
// BN apply + optional residual (channel-duplicating concat when res_c < C)
// + leaky-ReLU(0.2). Writes f32 result and its bf16 shadow (next conv input).
// ---------------------------------------------------------------------------
__global__ __launch_bounds__(256)
void bn_apply(const float* __restrict__ z, const float* __restrict__ sums,
              const float* __restrict__ g, const float* __restrict__ be,
              const float* __restrict__ res, int res_mask,
              float* __restrict__ out, __bf16* __restrict__ outb,
              int N, int Cshift)
{
    int C = 1 << Cshift;
    int idx = blockIdx.x * 256 + threadIdx.x;
    if (idx >= (N << Cshift)) return;
    int c = idx & (C - 1);
    float invN = 1.f / (float)N;
    float mu  = sums[c] * invN;
    float var = sums[1024 + c] * invN - mu * mu;
    float v = (z[idx] - mu) * rsqrtf(var + 1e-5f) * g[c] + be[c];
    if (res) {
        int r = idx >> Cshift;
        v += res[r * (res_mask + 1) + (c & res_mask)];
    }
    v = (v > 0.f) ? v : 0.2f * v;
    out[idx]  = v;
    outb[idx] = f2bf(v);
}

// ---------------------------------------------------------------------------
// pool_max: faithful to the reference flat-gather reshape quirk:
//   flat = i*7*C + c*7 + j -> x[neigh[flat / C]][flat % C]   (C = power of 2)
// ---------------------------------------------------------------------------
__global__ __launch_bounds__(256)
void pool_max(const float* __restrict__ x, const int* __restrict__ neigh,
              float* __restrict__ out, __bf16* __restrict__ outb,
              int num, int Cshift)
{
    int C = 1 << Cshift;
    int idx = blockIdx.x * 256 + threadIdx.x;
    if (idx >= (num << Cshift)) return;
    int i = idx >> Cshift, c = idx & (C - 1);
    float m = -3.402823466e38f;
    int base = ((i * 7) << Cshift) + c * 7;
#pragma unroll
    for (int j = 0; j < 7; ++j) {
        int flat = base + j;
        int row = flat >> Cshift;
        int col = flat & (C - 1);
        float v = x[neigh[row] * C + col];
        m = (v > m) ? v : m;
    }
    out[idx]  = m;
    outb[idx] = f2bf(m);
}

// ---------------------------------------------------------------------------
// head: column mean over N rows (N=42), then 1024 -> 10 FC + sigmoid.
// ---------------------------------------------------------------------------
__global__ __launch_bounds__(256)
void head(const float* __restrict__ h, const float* __restrict__ fcW,
          const float* __restrict__ fcb, float* __restrict__ out, int N)
{
    __shared__ float mean[1024];
    for (int c = threadIdx.x; c < 1024; c += 256) {
        float s = 0.f;
        for (int r = 0; r < N; ++r) s += h[r * 1024 + c];
        mean[c] = s / (float)N;
    }
    __syncthreads();
    if (threadIdx.x < 10) {
        int o = threadIdx.x;
        float s = fcb[o];
        for (int c = 0; c < 1024; ++c) s += mean[c] * fcW[c * 10 + o];
        out[o] = 1.f / (1.f + expf(-s));
    }
}

// ---------------------------------------------------------------------------
// Host orchestration. Param leaves in jax canonical flatten order (sorted
// dict keys): bn1.b, bn1.g, conv1.W, conv1.b, fcW, fcb, then res[s][b] blocks
// of 8 leaves [bn1.b, bn1.g, bn2.b, bn2.g, conv1.W, conv1.b, conv2.W, conv2.b].
// ---------------------------------------------------------------------------
static inline int ctz_i(int v) { int s = 0; while (!(v & 1)) { v >>= 1; ++s; } return s; }

struct BufT { float* f; __bf16* b; };

extern "C" void kernel_launch(void* const* d_in, const int* in_sizes, int n_in,
                              void* d_out, int out_size, void* d_ws, size_t ws_size,
                              hipStream_t stream)
{
    (void)in_sizes; (void)n_in; (void)out_size; (void)ws_size;

    const float* x = (const float*)d_in[0];
    const int* ngh[6];
    for (int i = 0; i < 6; ++i) ngh[i] = (const int*)d_in[1 + i];

    auto P = [&](int i) { return (const float*)d_in[i]; };
    const float* bn1_b   = P(7);
    const float* bn1_g   = P(8);
    const float* conv1_W = P(9);
    const float* conv1_b = P(10);
    const float* fcW     = P(11);
    const float* fcb     = P(12);
    auto blkp = [&](int s, int b, int leaf) { return P(13 + (s * 3 + b) * 8 + leaf); };

    char* ws = (char*)d_ws;
    const size_t FBUF = (size_t)12 << 20;     // f32 buffers
    const size_t SBUF = (size_t)6  << 20;     // bf16 shadows
    BufT B[4];
    for (int i = 0; i < 4; ++i) {
        B[i].f = (float*)(ws + (size_t)i * FBUF);
        B[i].b = (__bf16*)(ws + 4 * FBUF + (size_t)i * SBUF);
    }
    __bf16* xb   = (__bf16*)(ws + 4 * FBUF + 4 * SBUF);              // 6 MB
    __bf16* Wf   = (__bf16*)(ws + 4 * FBUF + 5 * SBUF);              // 16 MB
    float*  sums = (float*) (ws + 4 * FBUF + 5 * SBUF + ((size_t)16 << 20));

    auto conv = [&](const __bf16* inb, const int* ne, const float* Wt,
                    const float* bs, float* z, int N, int cin, int cout) {
        int K = 7 * cin;
        int tot = K * cout;
        hipLaunchKernelGGL(cvt_w, dim3((tot + 255) / 256), dim3(256), 0, stream,
                           Wt, Wf, K, cout);
        dim3 g((N + 31) / 32, (cout + 127) / 128);
        hipLaunchKernelGGL(conv_wmma, g, dim3(256), 0, stream,
                           inb, ne, Wf, bs, z, N, ctz_i(cin), cout);
    };
    auto bn = [&](const float* z, const float* gg, const float* bb,
                  const float* res, int res_c, BufT out, int N, int C) {
        hipMemsetAsync(sums, 0, 2 * 1024 * sizeof(float), stream);
        hipLaunchKernelGGL(bn_stats, dim3((N + 127) / 128), dim3(256), 0, stream,
                           z, sums, N, C);
        int tot = N * C;
        hipLaunchKernelGGL(bn_apply, dim3((tot + 255) / 256), dim3(256), 0, stream,
                           z, sums, gg, bb, res, res_c - 1, out.f, out.b,
                           N, ctz_i(C));
    };
    auto pool = [&](const float* in, const int* ne, BufT out, int num, int C) {
        int tot = num * C;
        hipLaunchKernelGGL(pool_max, dim3((tot + 255) / 256), dim3(256), 0, stream,
                           in, ne, out.f, out.b, num, ctz_i(C));
    };

    // conv1 + bn + lrelu at level 40962
    int N = 40962;
    hipLaunchKernelGGL(cvt_x, dim3((N * 64 + 255) / 256), dim3(256), 0, stream,
                       x, xb, N * 64);
    conv(xb, ngh[0], conv1_W, conv1_b, B[1].f, N, 64, 64);
    bn(B[1].f, bn1_g, bn1_b, nullptr, 1, B[0], N, 64);

    // pool -> 10242
    int num = (N + 6) / 4;
    pool(B[0].f, ngh[0], B[2], num, 64);
    N = num;

    BufT act = B[2], zb = B[1], hb = B[0], nxt = B[3];

    const int scin[5]  = {64, 64, 128, 256, 512};
    const int scout[5] = {64, 128, 256, 512, 1024};

    for (int s = 0; s < 5; ++s) {
        if (s > 0) {
            num = (N + 6) / 4;
            pool(act.f, ngh[s], nxt, num, scin[s]);
            BufT t = act; act = nxt; nxt = t;
            N = num;
        }
        for (int b = 0; b < 3; ++b) {
            int cin  = (b == 0) ? scin[s] : scout[s];
            int cout = scout[s];
            // h1 = lrelu(bn1(conv1(act)))
            conv(act.b, ngh[s + 1], blkp(s, b, 4), blkp(s, b, 5), zb.f, N, cin, cout);
            bn(zb.f, blkp(s, b, 1), blkp(s, b, 0), nullptr, 1, hb, N, cout);
            // out = lrelu(bn2(conv2(h1)) + residual)  (channel-dup if cin<cout)
            conv(hb.b, ngh[s + 1], blkp(s, b, 6), blkp(s, b, 7), zb.f, N, cout, cout);
            bn(zb.f, blkp(s, b, 3), blkp(s, b, 2), act.f, cin, nxt, N, cout);
            BufT t = act; act = nxt; nxt = t;
        }
    }

    // head: mean over 42 nodes -> FC(1024,10) -> sigmoid
    hipLaunchKernelGGL(head, dim3(1), dim3(256), 0, stream,
                       act.f, fcW, fcb, (float*)d_out, N);
}
